// CustomDCNv2_45629732552757
// MI455X (gfx1250) — matrix operand order repaired and checked by reading, not compiled, not run on hardware
//
#include <hip/hip_runtime.h>
#include <hip/hip_bf16.h>

// ---------------------------------------------------------------------------
// DCNv2 fused: bilinear-im2col + f16 WMMA GEMM for gfx1250 (MI455X).
//   D[Co=256, P=16384] = W[256, K=2304] x S[2304, P]
// K ordering k-major (kidx = k*256 + c): each 32-wide K chunk has a single
// 3x3 tap -> one bilinear stencil per (pixel, chunk), cached in LDS.
// Software-pipelined with a manually unrolled 2-phase steady state so the
// A-fragment double buffer stays in registers (no dynamic indexing -> no
// scratch spills): B tile ping-pongs in LDS, next chunk's L2 gathers and
// A loads overlap the current chunk's 8 v_wmma_f32_16x16x32_f16.
// ---------------------------------------------------------------------------

typedef __attribute__((ext_vector_type(16))) _Float16 v16h;
typedef __attribute__((ext_vector_type(8)))  _Float16 v8h;
typedef __attribute__((ext_vector_type(8)))  float    v8f;

#define CIN    256
#define COUT   256
#define HW     64
#define KTAPS  9
#define NCHUNK 72           // 2304/32 (even -> clean 2-phase pipeline)
#define BROW   40           // padded LDS row stride (halves) for B tile

// ---------------------------------------------------------------------------
// Pass 1: weight f32->f16, pre-swizzled into the per-lane A-fragment stream
// for V_WMMA_F32_16X16X32_F16 (contiguous K order k = 16*(lane/16) + elem, same as the B tile):
//   lane<16 : elems 0..7 = K 0..7  , elems 8..15 = K 8..15   (row M = lane)
//   lane>=16: elems 0..7 = K 16..23, elems 8..15 = K 24..31  (row M = lane-16)
// Flat: wf[(((mt*72 + kc)*2 + seg)*32 + lane)*8 + h]  -> two coalesced b128
// loads per fragment in the GEMM.
// ---------------------------------------------------------------------------
__global__ void dcn_weight_swizzle(const float* __restrict__ w,
                                   _Float16* __restrict__ wf) {
  int i = blockIdx.x * 256 + threadIdx.x;          // < 589824 = 2304 blocks
  if (i >= 16 * NCHUNK * 2 * 32 * 8) return;
  int h    = i & 7;
  int lane = (i >> 3) & 31;
  int seg  = (i >> 8) & 1;
  int kc   = (i >> 9) % NCHUNK;
  int mt   = i / (NCHUNK * 512);
  int M    = mt * 16 + (lane & 15);
  int kloc = ((lane >> 4) << 4) + (seg << 3) + h;  // K within 32-chunk
  int kg   = kc * 32 + kloc;                       // k-major global K index
  int c    = kg & 255;
  int k    = kg >> 8;                              // 3x3 tap
  wf[i] = (_Float16)w[(M * CIN + c) * KTAPS + k];
}

// ---------------------------------------------------------------------------
// Pass 2: fused sampling + GEMM. One WG per output row (n,h): 64 px x 256 Co.
// ---------------------------------------------------------------------------
__global__ void __launch_bounds__(256)
dcn_main(const float* __restrict__ x,      // [N,256,64,64]
         const float* __restrict__ off,    // [N,18,64,64]
         const float* __restrict__ mask,   // [N,9,64,64]
         const _Float16* __restrict__ wf,  // pre-swizzled weight
         const float* __restrict__ bias,   // [256]
         float* __restrict__ out) {        // [N,256,64,64]
  __shared__ __align__(16) int      SI[KTAPS][HW][4];     // clipped corner idx
  __shared__ __align__(16) float    SW[KTAPS][HW][4];     // mask*bilin weights
  __shared__ __align__(16) _Float16 ST[2][HW * BROW];     // ping-pong B tile

  const int t  = threadIdx.x;
  const int wg = blockIdx.x;            // 0..255
  const int n  = wg >> 6;
  const int h  = wg & 63;

  // --- per-(pixel, tap) bilinear parameters, computed once per WG ---
  for (int e = t; e < KTAPS * HW; e += 256) {
    int k = e >> 6, p = e & 63;
    int ki = k / 3, kj = k - ki * 3;
    float dy = off[(((n * 18) + 2 * k    ) * HW + h) * HW + p];
    float dx = off[(((n * 18) + 2 * k + 1) * HW + h) * HW + p];
    float m  = mask[((n * KTAPS + k) * HW + h) * HW + p];
    float py = dy + (float)(h - 1 + ki);
    float px = dx + (float)(p - 1 + kj);
    float y0f = floorf(py), x0f = floorf(px);
    int y0 = (int)y0f, x0 = (int)x0f, y1 = y0 + 1, x1 = x0 + 1;
    float wy1 = py - y0f, wx1 = px - x0f;
    float wy0 = 1.f - wy1, wx0 = 1.f - wx1;
    bool vy0 = (y0 >= 0) & (y0 < HW), vy1 = (y1 >= 0) & (y1 < HW);
    bool vx0 = (x0 >= 0) & (x0 < HW), vx1 = (x1 >= 0) & (x1 < HW);
    int cy0 = min(max(y0, 0), HW - 1), cy1 = min(max(y1, 0), HW - 1);
    int cx0 = min(max(x0, 0), HW - 1), cx1 = min(max(x1, 0), HW - 1);
    SI[k][p][0] = cy0 * HW + cx0;  SW[k][p][0] = (vy0 && vx0) ? m * wy0 * wx0 : 0.f;
    SI[k][p][1] = cy0 * HW + cx1;  SW[k][p][1] = (vy0 && vx1) ? m * wy0 * wx1 : 0.f;
    SI[k][p][2] = cy1 * HW + cx0;  SW[k][p][2] = (vy1 && vx0) ? m * wy1 * wx0 : 0.f;
    SI[k][p][3] = cy1 * HW + cx1;  SW[k][p][3] = (vy1 && vx1) ? m * wy1 * wx1 : 0.f;
  }

  const int wave = t >> 5, lane = t & 31;
  const int p_s  = t & 63, cg = t >> 6;          // sampling role: pixel, ch-grp
  const float* xn = x + (size_t)n * CIN * (HW * HW);
  const int koff = (lane >> 4) << 4;             // B: lanes<16 K0..15, else K16..31
  const int col  = lane & 15;

  union AF { v16h v; v8h h2[2]; };
  AF aA[2], aB[2];                               // named ping-pong A fragments
  v8f acc[4][2] = {};                            // [nt][mtile]

  // sample one 64px x 32ch chunk into an LDS B-tile buffer
  auto sampleChunk = [&](int kc, _Float16* dst) {
    int k = kc >> 3;
    int cbase = (kc & 7) << 5;
    int4   i4 = *(const int4*)  SI[k][p_s];
    float4 w4 = *(const float4*)SW[k][p_s];
    const float* bp = xn + (size_t)(cbase + (cg << 3)) * (HW * HW);
    v8h hv;
#pragma unroll
    for (int cc = 0; cc < 8; ++cc) {
      const float* q = bp + cc * (HW * HW);
      hv[cc] = (_Float16)(w4.x * q[i4.x] + w4.y * q[i4.y] +
                          w4.z * q[i4.z] + w4.w * q[i4.w]);
    }
    *(v8h*)&dst[p_s * BROW + (cg << 3)] = hv;
  };
  // load this wave's two A fragments (4 coalesced b128 from L2-resident wf)
  auto loadA = [&](int kc, AF* d) {
    const _Float16* a0p = wf + ((size_t)(wave * 2) * NCHUNK + kc) * 512 + lane * 8;
    const _Float16* a1p = a0p + (size_t)NCHUNK * 512;
    d[0].h2[0] = *(const v8h*)a0p;  d[0].h2[1] = *(const v8h*)(a0p + 256);
    d[1].h2[0] = *(const v8h*)a1p;  d[1].h2[1] = *(const v8h*)(a1p + 256);
  };
  // B fragments + 8 WMMAs for one chunk (EXEC all-ones, uniform flow)
  auto mma = [&](const _Float16* STcur, AF* af) {
    v16h b[4];
#pragma unroll
    for (int nt = 0; nt < 4; ++nt)
      b[nt] = *(const v16h*)&STcur[(nt * 16 + col) * BROW + koff];
#pragma unroll
    for (int nt = 0; nt < 4; ++nt) {
      acc[nt][0] = __builtin_amdgcn_wmma_f32_16x16x32_f16(
          false, af[0].v, false, b[nt], (short)0, acc[nt][0], false, false);
      acc[nt][1] = __builtin_amdgcn_wmma_f32_16x16x32_f16(
          false, af[1].v, false, b[nt], (short)0, acc[nt][1], false, false);
    }
  };

  // prologue: chunk 0 tile + A fragments
  sampleChunk(0, ST[0]);
  loadA(0, aA);
  __syncthreads();

#pragma unroll 1
  for (int kc = 0; kc < NCHUNK; kc += 2) {
    // even phase: consume ST[0]/aA, prefetch kc+1 -> ST[1]/aB (kc+1 <= 71)
    sampleChunk(kc + 1, ST[1]);
    loadA(kc + 1, aB);
    mma(ST[0], aA);
    __syncthreads();

    // odd phase: consume ST[1]/aB, prefetch kc+2 -> ST[0]/aA (guarded)
    if (kc + 2 < NCHUNK) {
      sampleChunk(kc + 2, ST[0]);
      loadA(kc + 2, aA);
    }
    mma(ST[1], aB);
    __syncthreads();
  }

  // --- epilogue: D VGPR r -> (M = r + 8*(lane>=16), N = lane%16); add bias ---
  const int nlo = lane & 15, hi = lane >> 4;
#pragma unroll
  for (int mtl = 0; mtl < 2; ++mtl) {
#pragma unroll
    for (int r = 0; r < 8; ++r) {
      int co = wave * 32 + mtl * 16 + r + hi * 8;
      float bv = bias[co];
#pragma unroll
      for (int nt = 0; nt < 4; ++nt) {
        int wo = nt * 16 + nlo;
        out[(((size_t)n * COUT + co) * HW + h) * HW + wo] = acc[nt][mtl][r] + bv;
      }
    }
  }
}

extern "C" void kernel_launch(void* const* d_in, const int* in_sizes, int n_in,
                              void* d_out, int out_size, void* d_ws, size_t ws_size,
                              hipStream_t stream) {
  const float* x      = (const float*)d_in[0];
  const float* offset = (const float*)d_in[1];
  const float* mask   = (const float*)d_in[2];
  const float* weight = (const float*)d_in[3];
  const float* bias   = (const float*)d_in[4];
  // d_in[5..8] = stride/padding/dilation/dg scalars: fixed (1,1,1,1) config.
  _Float16* wf = (_Float16*)d_ws;                 // 589824 halves = 1.18 MB

  dcn_weight_swizzle<<<2304, 256, 0, stream>>>(weight, wf);
  dcn_main<<<256, 256, 0, stream>>>(x, offset, mask, wf, bias, (float*)d_out);
}
